// AdditiveAttention_50809463112051
// MI455X (gfx1250) — matrix operand order repaired and checked by reading, not compiled, not run on hardware
//
#include <hip/hip_runtime.h>
#include <hip/hip_bf16.h>

#define B_ 32
#define T_ 2048
#define D_ 1024
#define U_ 1024

typedef __attribute__((ext_vector_type(16))) __bf16 bf16x16;
typedef __attribute__((ext_vector_type(8)))  float  f32x8;

__device__ __forceinline__ unsigned short f2bf_rne(float f) {
  unsigned u = __float_as_uint(f);
  unsigned r = 0x7FFFu + ((u >> 16) & 1u);
  u += r;
  return (unsigned short)(u >> 16);
}

// ---------------------------------------------------------------------------
// W1 [D,U] fp32 row-major  ->  bf16, pre-swizzled into WMMA B-fragment order.
// This file's k order (A matches): lanes 0-15 hold K=0..15 (2 per VGPR), lanes 16-31
// hold K=16..31; N = lane & 15.  Fragment storage: 32 lanes x 16 bf16,
// indexed [(ntile*32 + kstep)*32 + lane]*16 + e  so each lane loads 32
// contiguous bytes.
// ---------------------------------------------------------------------------
__global__ void k_w1_swizzle(const float* __restrict__ W1,
                             unsigned short* __restrict__ w1bf) {
  int idx = blockIdx.x * 256 + threadIdx.x;   // 0 .. D*U-1
  int k = idx >> 10;                          // row (D index)
  int n = idx & 1023;                         // col (U index)
  int kstep = k >> 5;
  int kk    = k & 31;
  int h  = (kk >> 4) & 1;
  int j  = (kk & 15) >> 1;
  int jj = kk & 1;
  int lane  = h * 16 + (n & 15);
  int ntile = n >> 4;
  int e = 2 * j + jj;
  int dst = ((ntile * 32 + kstep) * 32 + lane) * 16 + e;
  w1bf[dst] = f2bf_rne(W1[idx]);
}

// ---------------------------------------------------------------------------
// q_proj[b,u] = query[b,:] . W2[:,u] + b2[u]
// ---------------------------------------------------------------------------
__global__ void k_qproj(const float* __restrict__ query,
                        const float* __restrict__ W2,
                        const float* __restrict__ b2,
                        float* __restrict__ qproj) {
  int u = blockIdx.x * 256 + threadIdx.x;
  int b = blockIdx.y;
  const float* q = query + b * D_;
  float acc = b2[u];
  for (int d = 0; d < D_; ++d) acc += q[d] * W2[d * U_ + u];
  qproj[b * U_ + u] = acc;
}

// ---------------------------------------------------------------------------
// Fused score[b,t] = sum_u tanh(values[b,t,:].W1[:,u] + qproj[b,u] + b1[u])*V[u] + bv
// One block = 32 rows (one b, 32 consecutive t) = two 16-row A tiles.
// 8 waves; each wave covers a 128-wide U slice.  Each W1 B-fragment load is
// reused by 2 WMMAs (one per A tile) -> 32 FLOP per L2 byte.
// LDS: 64 KB A tile; the score-reduction buffer reuses the same LDS after the
// WMMA phase (barrier-separated).
// ---------------------------------------------------------------------------
__global__ void __launch_bounds__(256) k_scores(
    const float* __restrict__ values, const unsigned short* __restrict__ w1bf,
    const float* __restrict__ qproj, const float* __restrict__ b1,
    const float* __restrict__ V, const float* __restrict__ bv,
    float* __restrict__ scores) {
  __shared__ __align__(32) unsigned short lds_a[32 * 1024];  // 64 KB

  const int tid   = threadIdx.x;
  const int ttile = blockIdx.x;   // 0..63  (32 rows each)
  const int b     = blockIdx.y;   // 0..31

  // ---- Phase 1: 32x1024 fp32 values rows -> bf16 A fragments in LDS ----
  // A fragment, contiguous k (same as B): lanes0-15: M=lane, VGPR0..7=K0..15
  //                          lanes16-31: K16..31.
  // Two A tiles: rows 0-15 at short-offset 0, rows 16-31 at 16384.
  const float4* src = (const float4*)(values + (size_t)(b * T_ + ttile * 32) * D_);
  unsigned int* lds_u32 = (unsigned int*)lds_a;
  for (int c = 0; c < 32; ++c) {
    int f4 = c * 256 + tid;            // 0..8191, coalesced
    float4 v = src[f4];
    int f = f4 * 4;
    int m = f >> 10;                   // row in 32-row tile
    int kbase = f & 1023;
    int tile_m = m >> 4;
    int mrow   = m & 15;
    float vv[4] = {v.x, v.y, v.z, v.w};
#pragma unroll
    for (int p = 0; p < 4; p += 2) {   // pairs of consecutive K share a dword
      int k0 = kbase + p;
      int kk = k0 & 31;
      int kstep = k0 >> 5;
      int h = (kk >> 4) & 1;
      int j = (kk & 15) >> 1;
      int lane = h * 16 + mrow;
      int e = 2 * j;                   // jj = 0 (k0 even)
      int dst = tile_m * 16384 + (kstep * 32 + lane) * 16 + e;
      unsigned int pack = (unsigned int)f2bf_rne(vv[p]) |
                          ((unsigned int)f2bf_rne(vv[p + 1]) << 16);
      lds_u32[dst >> 1] = pack;
    }
  }
  __syncthreads();

  const int wave = tid >> 5;
  const int lane = tid & 31;

  f32x8 acc[2][8];
#pragma unroll
  for (int tm = 0; tm < 2; ++tm)
#pragma unroll
    for (int nt = 0; nt < 8; ++nt) acc[tm][nt] = {};

  const bf16x16* af  = (const bf16x16*)lds_a;        // tile 1 at +1024 frags
  const bf16x16* bfm = (const bf16x16*)w1bf;

  // ---- Phase 2: K loop; each B fragment feeds 2 WMMAs ----
  for (int kstep = 0; kstep < 32; ++kstep) {
    bf16x16 a0 = af[kstep * 32 + lane];
    bf16x16 a1 = af[1024 + kstep * 32 + lane];
#pragma unroll
    for (int nt = 0; nt < 8; ++nt) {
      int ntile = wave * 8 + nt;
      bf16x16 bb = bfm[((size_t)(ntile * 32 + kstep) * 32) + lane];
      acc[0][nt] = __builtin_amdgcn_wmma_f32_16x16x32_bf16(
          false, a0, false, bb, (short)0, acc[0][nt], false, false);
      acc[1][nt] = __builtin_amdgcn_wmma_f32_16x16x32_bf16(
          false, a1, false, bb, (short)0, acc[1][nt], false, false);
    }
  }

  // ---- Epilogue: tanh + dot with V, reduce over U ----
  // C layout: element i of lane l is (M = i + 8*(l>=16), N = l&15).
  float sacc[2][8];
#pragma unroll
  for (int tm = 0; tm < 2; ++tm)
#pragma unroll
    for (int i = 0; i < 8; ++i) sacc[tm][i] = 0.f;
#pragma unroll
  for (int nt = 0; nt < 8; ++nt) {
    int u = ((wave * 8 + nt) << 4) + (lane & 15);
    float qb = qproj[b * U_ + u] + b1[u];
    float vu = V[u];
#pragma unroll
    for (int tm = 0; tm < 2; ++tm)
#pragma unroll
      for (int i = 0; i < 8; ++i)
        sacc[tm][i] += tanhf(acc[tm][nt][i] + qb) * vu;
  }

  __syncthreads();                     // all waves done reading lds_a
  float* lds_red = (float*)lds_a;      // reuse LDS: 2*256*8 floats = 16 KB
#pragma unroll
  for (int tm = 0; tm < 2; ++tm)
#pragma unroll
    for (int i = 0; i < 8; ++i)
      lds_red[(tm * 256 + tid) * 8 + i] = sacc[tm][i];
  __syncthreads();

  if (tid < 32) {                      // one thread per row m
    int m = tid;
    int tm = m >> 4;
    int mm = m & 15;
    int i = mm & 7;
    int h = mm >> 3;
    float s = 0.f;
    for (int w = 0; w < 8; ++w)
      for (int l = 0; l < 16; ++l)
        s += lds_red[(tm * 256 + w * 32 + h * 16 + l) * 8 + i];
    scores[b * T_ + ttile * 32 + m] = s + bv[0];
  }
}

// ---------------------------------------------------------------------------
// Softmax over T per batch; writes attention weights to output.
// ---------------------------------------------------------------------------
__global__ void k_softmax(const float* __restrict__ scores,
                          float* __restrict__ weights) {
  __shared__ float red[256];
  int b = blockIdx.x, tid = threadIdx.x;
  float s[8];
  float mx = -3.4e38f;
#pragma unroll
  for (int j = 0; j < 8; ++j) {
    s[j] = scores[b * T_ + j * 256 + tid];
    mx = fmaxf(mx, s[j]);
  }
  red[tid] = mx; __syncthreads();
  for (int off = 128; off > 0; off >>= 1) {
    if (tid < off) red[tid] = fmaxf(red[tid], red[tid + off]);
    __syncthreads();
  }
  mx = red[0]; __syncthreads();
  float sum = 0.f;
#pragma unroll
  for (int j = 0; j < 8; ++j) { s[j] = __expf(s[j] - mx); sum += s[j]; }
  red[tid] = sum; __syncthreads();
  for (int off = 128; off > 0; off >>= 1) {
    if (tid < off) red[tid] += red[tid + off];
    __syncthreads();
  }
  float inv = 1.f / red[0];
#pragma unroll
  for (int j = 0; j < 8; ++j) weights[b * T_ + j * 256 + tid] = s[j] * inv;
}

// ---------------------------------------------------------------------------
// context partials: 8 deterministic T-chunks per batch.
// ---------------------------------------------------------------------------
__global__ void k_ctx_partial(const float* __restrict__ values,
                              const float* __restrict__ weights,
                              float* __restrict__ partial) {
  int tch = blockIdx.x;   // 0..7
  int b   = blockIdx.y;   // 0..31
  int tid = threadIdx.x;  // d4 index (4 floats each -> D=1024)
  const float4* v4 = (const float4*)values;
  float4 acc = {0.f, 0.f, 0.f, 0.f};
  for (int it = 0; it < 256; ++it) {
    int t = tch * 256 + it;
    float w = weights[b * T_ + t];
    float4 v = v4[(size_t)(b * T_ + t) * (D_ / 4) + tid];
    acc.x += w * v.x; acc.y += w * v.y; acc.z += w * v.z; acc.w += w * v.w;
  }
  ((float4*)partial)[(size_t)(tch * B_ + b) * (D_ / 4) + tid] = acc;
}

__global__ void k_ctx_reduce(const float* __restrict__ partial,
                             float* __restrict__ ctx) {
  int idx = blockIdx.x * 256 + threadIdx.x;  // 0..B*D-1
  float s = 0.f;
  for (int tch = 0; tch < 8; ++tch) s += partial[tch * (B_ * D_) + idx];
  ctx[idx] = s;
}

// ---------------------------------------------------------------------------
extern "C" void kernel_launch(void* const* d_in, const int* in_sizes, int n_in,
                              void* d_out, int out_size, void* d_ws, size_t ws_size,
                              hipStream_t stream) {
  (void)in_sizes; (void)n_in; (void)out_size; (void)ws_size;
  const float* query  = (const float*)d_in[0];
  const float* values = (const float*)d_in[1];
  const float* W1     = (const float*)d_in[2];
  const float* b1     = (const float*)d_in[3];
  const float* W2     = (const float*)d_in[4];
  const float* b2     = (const float*)d_in[5];
  const float* V      = (const float*)d_in[6];
  const float* bv     = (const float*)d_in[7];

  float* out     = (float*)d_out;
  float* ctx_out = out;               // [B, D]
  float* wts_out = out + B_ * D_;     // [B, T] (attention weights)

  char* ws = (char*)d_ws;
  unsigned short* w1bf = (unsigned short*)ws;                                   // 2 MB
  float* qproj   = (float*)(ws + (size_t)2 * 1024 * 1024);                      // 128 KB
  float* scores  = (float*)(ws + (size_t)2 * 1024 * 1024 + 131072);             // 256 KB
  float* partial = (float*)(ws + (size_t)2 * 1024 * 1024 + 131072 + 262144);    // 1 MB

  k_w1_swizzle<<<dim3((D_ * U_) / 256), dim3(256), 0, stream>>>(W1, w1bf);
  k_qproj    <<<dim3(U_ / 256, B_),    dim3(256), 0, stream>>>(query, W2, b2, qproj);
  k_scores   <<<dim3(T_ / 32, B_),     dim3(256), 0, stream>>>(values, w1bf, qproj, b1, V, bv, scores);
  k_softmax  <<<dim3(B_),              dim3(256), 0, stream>>>(scores, wts_out);
  k_ctx_partial<<<dim3(8, B_),         dim3(256), 0, stream>>>(values, wts_out, partial);
  k_ctx_reduce <<<dim3((B_ * D_) / 256), dim3(256), 0, stream>>>(partial, ctx_out);
}
